// MultiHeadAttention_43963285241945
// MI455X (gfx1250) — compile-verified, matrix operand order checked by reading, not run on hardware
//
#include <hip/hip_runtime.h>
#include <hip/hip_bf16.h>

#define D_MODEL 1024
#define SEQ     2048
#define NB      4
#define NH      16
#define HEAD    64

// LDS strides (bf16 elements). All rows 16B-aligned; 88/40 give conflict-free
// ds_load_b128 fragment reads (r*stride/4 mod 64 all distinct).
#define KLD 88
#define VLD 88
#define PLD 80
#define GLD 40

typedef __attribute__((ext_vector_type(16))) __bf16 v16bf;
typedef __attribute__((ext_vector_type(8)))  float  v8f;
typedef unsigned int v4u __attribute__((ext_vector_type(4)));
typedef int          v4i __attribute__((ext_vector_type(4)));
typedef int          v8i __attribute__((ext_vector_type(8)));

#if defined(__AMDGCN__) && __has_builtin(__builtin_amdgcn_tensor_load_to_lds)
#define USE_TDM 1
#else
#define USE_TDM 0
#endif

// ---------------------------------------------------------------------------
__device__ __forceinline__ v8f wmma_bf16(v16bf a, v16bf b, v8f c) {
  return __builtin_amdgcn_wmma_f32_16x16x32_bf16(
      false, a, false, b, (short)0, c, false, false);
}

// Fragment from a row-major [16 x >=32] bf16 tile whose "rows" are the lane
// index (M for A, N for B-transposed). Two aligned 16-byte loads per lane.
__device__ __forceinline__ v16bf load_frag(const __bf16* src, int ld, int lane) {
  const int r  = lane & 15;
  const int kb = (lane >> 4) << 3;
  const __bf16* p = src + r * ld + kb;
  union { v16bf v; uint4 u[2]; } f;
  f.u[0] = *(const uint4*)(p);
  f.u[1] = *(const uint4*)(p + 16);
  return f.v;
}

#if USE_TDM
// 2-D bf16 tile load via Tensor Data Mover (ISA ch.8 D# layout), 6-arg builtin.
// rows = tile_dim1, cols = tile_dim0 (elements), row_stride in elements.
__device__ __forceinline__ void tdm_load_2d_bf16(unsigned lds_off,
                                                 const void* gaddr,
                                                 unsigned rows, unsigned cols,
                                                 unsigned row_stride,
                                                 int pad_int_code,
                                                 int pad_amt_code) {
  unsigned long long ga = (unsigned long long)gaddr;
  v4u g0;
  g0.x = 1u;                                        // count=1, user mode
  g0.y = lds_off;                                   // lds_addr (bytes)
  g0.z = (unsigned)(ga & 0xffffffffu);              // global_addr[31:0]
  g0.w = (unsigned)((ga >> 32) & 0x01ffffffu) | (2u << 30);  // addr hi | type=2
  unsigned dw0 = (1u << 16);                        // data_size = 2 bytes
  if (pad_amt_code >= 0)
    dw0 |= (1u << 20) | ((unsigned)pad_int_code << 22) |
           ((unsigned)pad_amt_code << 25);
  v8i g1;
  g1[0] = (int)dw0;
  g1[1] = (int)((cols & 0xffffu) << 16);            // tensor_dim0[15:0]
  g1[2] = (int)((cols >> 16) | ((rows & 0xffffu) << 16));  // dim0 hi | dim1 lo
  g1[3] = (int)((rows >> 16) | ((cols & 0xffffu) << 16));  // dim1 hi | tile_dim0
  g1[4] = (int)(rows & 0xffffu);                    // tile_dim1 (tile_dim2 = 0)
  g1[5] = (int)row_stride;                          // tensor_dim0_stride[31:0]
  g1[6] = 0;
  g1[7] = 0;
  v4i z4 = {0, 0, 0, 0};
  v8i z8 = {0, 0, 0, 0, 0, 0, 0, 0};
  __builtin_amdgcn_tensor_load_to_lds(g0, g1, z4, z4, z8, 0);
}
#endif

// ---------------------------------------------------------------------------
// Kernel 1: per-head QKV projection.  X:[B*S,1024]f32  W:[H,1024,64]f32
//   out:[B,H,S,64]bf16 (scaled by outScale).
// 128 threads / 4 waves; block tile 128(M) x 64(N); wave tile 32 x 64; K=32.
// ---------------------------------------------------------------------------
__global__ __launch_bounds__(128) void qkv_proj_kernel(
    const float* __restrict__ X, const float* __restrict__ W,
    __bf16* __restrict__ out, float outScale) {
  __shared__ __align__(16) __bf16 aT[128 * GLD];    // [m][k], 128x32
  __shared__ __align__(16) __bf16 bTt[64 * GLD];    // transposed [n][k], 64x32
  const int lane = threadIdx.x & 31;
  const int wave = threadIdx.x >> 5;
  const int row0 = blockIdx.x * 128;
  const int h    = blockIdx.y;

  v8f acc[2][4] = {};

  for (int k0 = 0; k0 < D_MODEL; k0 += 32) {
    __syncthreads();
    // A tile: one 32-element row per thread (fp32 -> bf16, packs to cvt_pk,
    // 16B-aligned contiguous LDS stores).
    {
      const float* xr = X + (size_t)(row0 + threadIdx.x) * D_MODEL + k0;
      __bf16* ar = &aT[threadIdx.x * GLD];
#pragma unroll
      for (int j = 0; j < 32; ++j) ar[j] = (__bf16)xr[j];
    }
    // B tile transposed: bTt[n][k] = W[h][k0+k][n] (reads coalesce across
    // threads along n; LDS writes contiguous along k).
    {
      int n  = threadIdx.x & 63;
      int kh = (threadIdx.x >> 6) << 4;
#pragma unroll
      for (int i = 0; i < 16; ++i)
        bTt[n * GLD + kh + i] =
            (__bf16)W[((size_t)h * D_MODEL + (k0 + kh + i)) * HEAD + n];
    }
    __syncthreads();

    v16bf a0 = load_frag(&aT[(wave * 32) * GLD], GLD, lane);
    v16bf a1 = load_frag(&aT[(wave * 32 + 16) * GLD], GLD, lane);
#pragma unroll
    for (int nt = 0; nt < 4; ++nt) {
      v16bf b = load_frag(&bTt[(nt * 16) * GLD], GLD, lane);
      acc[0][nt] = wmma_bf16(a0, b, acc[0][nt]);
      acc[1][nt] = wmma_bf16(a1, b, acc[1][nt]);
    }
  }

  const int nlo = lane & 15;
#pragma unroll
  for (int rt = 0; rt < 2; ++rt) {
    const int rbase = row0 + wave * 32 + rt * 16 + ((lane >> 4) << 3);
#pragma unroll
    for (int nt = 0; nt < 4; ++nt) {
#pragma unroll
      for (int r = 0; r < 8; ++r) {
        int m  = rbase + r;
        int bb = m >> 11, s = m & (SEQ - 1);
        out[(((size_t)bb * NH + h) * SEQ + s) * HEAD + nt * 16 + nlo] =
            (__bf16)(acc[rt][nt][r] * outScale);
      }
    }
  }
}

// ---------------------------------------------------------------------------
// Kernel 2: flash attention, double-buffered K(TDM)/V staging.
//   q,k,v:[B,H,S,64]bf16 (q pre-scaled) -> ctx:[B,S,1024]bf16
// ---------------------------------------------------------------------------
__global__ __launch_bounds__(128) void flash_attn_kernel(
    const __bf16* __restrict__ q, const __bf16* __restrict__ k,
    const __bf16* __restrict__ v, __bf16* __restrict__ ctx) {
  __shared__ __align__(16) __bf16 kT[2][64 * KLD];  // row-major [key][dim]
  __shared__ __align__(16) __bf16 vTt[2][64 * VLD]; // transposed [dim][key]
  __shared__ __align__(16) __bf16 pT[4][16 * PLD];  // per-wave P scratch

  const int    lane  = threadIdx.x & 31;
  const int    wave  = threadIdx.x >> 5;
  const int    qrow0 = blockIdx.x * 64;
  const int    bh    = blockIdx.y;
  const size_t base  = (size_t)bh * SEQ * HEAD;

  // Stage key-tile kt into buffer buf: K via TDM (padded to stride 88),
  // V transposed via packed b32 stores (two key rows x 16 dims per thread).
  auto stage = [&](int kt, int buf) {
#if USE_TDM
    if (wave == 0)
      tdm_load_2d_bf16((unsigned)(unsigned long long)&kT[buf][0],
                       k + base + (size_t)kt * HEAD, 64u, 64u, 64u, 4, 11);
#else
    {
      const uint4* kg  = (const uint4*)(k + base + (size_t)kt * HEAD);
      uint4*       kTu = (uint4*)&kT[buf][0];       // 88 bf16 = 11 uint4 / row
      for (int c = threadIdx.x; c < 512; c += 128)
        kTu[(c >> 3) * 11 + (c & 7)] = kg[c];
    }
#endif
    const int kp  = (threadIdx.x >> 2) << 1;        // 0..62 step 2
    const int n0t = (threadIdx.x & 3) << 4;         // 0,16,32,48
    const __bf16* vr0 = v + base + (size_t)(kt + kp) * HEAD + n0t;
    const __bf16* vr1 = vr0 + HEAD;
#pragma unroll
    for (int j = 0; j < 16; ++j) {
      union { __bf16 h[2]; unsigned u; } pk;
      pk.h[0] = vr0[j];
      pk.h[1] = vr1[j];
      *(unsigned*)&vTt[buf][(n0t + j) * VLD + kp] = pk.u;
    }
  };

  const __bf16* qrow = q + base + (size_t)(qrow0 + wave * 16) * HEAD;
  v16bf aq0 = load_frag(qrow, HEAD, lane);
  v16bf aq1 = load_frag(qrow + 32, HEAD, lane);

  v8f   o[4] = {};
  float mrow[8], lrow[8];
#pragma unroll
  for (int r = 0; r < 8; ++r) { mrow[r] = -3.0e38f; lrow[r] = 0.0f; }

  // Prologue: stage tile 0, wait, barrier.
  stage(0, 0);
#if USE_TDM
  if (wave == 0) __builtin_amdgcn_s_wait_tensorcnt(0);
#endif
  __syncthreads();

  const int NTILE = SEQ / 64;
  for (int it = 0; it < NTILE; ++it) {
    const int cur = it & 1;
    // Prefetch next tile into the other buffer; overlaps compute below.
    if (it + 1 < NTILE) stage((it + 1) * 64, cur ^ 1);

    // S = Q @ K^T (B^T frag == row-major frag of the K tile).
    v8f s[4] = {};
#pragma unroll
    for (int nt = 0; nt < 4; ++nt) {
      v16bf b0 = load_frag(&kT[cur][(nt * 16) * KLD], KLD, lane);
      v16bf b1 = load_frag(&kT[cur][(nt * 16) * KLD + 32], KLD, lane);
      s[nt] = wmma_bf16(aq0, b0, s[nt]);
      s[nt] = wmma_bf16(aq1, b1, s[nt]);
    }

    // Online softmax (row r spans 16 lanes of one half-wave: xor masks 1..8).
    float mnew[8], alpha[8], rsum[8];
#pragma unroll
    for (int r = 0; r < 8; ++r) {
      float t = fmaxf(fmaxf(s[0][r], s[1][r]), fmaxf(s[2][r], s[3][r]));
#pragma unroll
      for (int off = 1; off <= 8; off <<= 1) t = fmaxf(t, __shfl_xor(t, off, 32));
      mnew[r]  = fmaxf(mrow[r], t);
      alpha[r] = __expf(mrow[r] - mnew[r]);
      mrow[r]  = mnew[r];
      rsum[r]  = 0.0f;
    }
#pragma unroll
    for (int nt = 0; nt < 4; ++nt)
#pragma unroll
      for (int r = 0; r < 8; ++r) {
        float p = __expf(s[nt][r] - mnew[r]);
        s[nt][r] = p;
        rsum[r] += p;
      }
#pragma unroll
    for (int r = 0; r < 8; ++r) {
#pragma unroll
      for (int off = 1; off <= 8; off <<= 1) rsum[r] += __shfl_xor(rsum[r], off, 32);
      lrow[r] = lrow[r] * alpha[r] + rsum[r];
    }
#pragma unroll
    for (int nt = 0; nt < 4; ++nt)
#pragma unroll
      for (int r = 0; r < 8; ++r) o[nt][r] *= alpha[r];

    // C-layout -> A-layout for P via per-wave LDS scratch (same-wave RAW).
    __bf16* pS = pT[wave];
    const int prow = (lane >> 4) << 3;
    const int pcol = lane & 15;
#pragma unroll
    for (int nt = 0; nt < 4; ++nt)
#pragma unroll
      for (int r = 0; r < 8; ++r)
        pS[(prow + r) * PLD + nt * 16 + pcol] = (__bf16)s[nt][r];

    v16bf ap0 = load_frag(pS, PLD, lane);
    v16bf ap1 = load_frag(pS + 32, PLD, lane);

    // O += P @ V (contiguous frags from the transposed V tile).
#pragma unroll
    for (int nt = 0; nt < 4; ++nt) {
      v16bf b0 = load_frag(&vTt[cur][(nt * 16) * VLD], VLD, lane);
      v16bf b1 = load_frag(&vTt[cur][(nt * 16) * VLD + 32], VLD, lane);
      o[nt] = wmma_bf16(ap0, b0, o[nt]);
      o[nt] = wmma_bf16(ap1, b1, o[nt]);
    }

    // Next tile's TDM + all waves' V stores must land before buffers flip.
#if USE_TDM
    if (wave == 0) __builtin_amdgcn_s_wait_tensorcnt(0);
#endif
    __syncthreads();
  }

  const int h     = bh & 15;
  const int bb    = bh >> 4;
  const int rbase = qrow0 + wave * 16 + ((lane >> 4) << 3);
  const int nlo   = lane & 15;
#pragma unroll
  for (int r = 0; r < 8; ++r) {
    float inv = 1.0f / lrow[r];
#pragma unroll
    for (int nt = 0; nt < 4; ++nt)
      ctx[((size_t)bb * SEQ + (rbase + r)) * D_MODEL + h * HEAD + nt * 16 + nlo] =
          (__bf16)(o[nt][r] * inv);
  }
}

// ---------------------------------------------------------------------------
// Kernel 3: out = ctx @ Wo^T + bo.  ctx:[B*S,1024]bf16, out fp32.
// Block tile 128(M) x 64(N); A tile staged via TDM (pad 16->20 dwords/row).
// ---------------------------------------------------------------------------
__global__ __launch_bounds__(128) void out_proj_kernel(
    const __bf16* __restrict__ ctx, const float* __restrict__ Wo,
    const float* __restrict__ bo, float* __restrict__ out) {
  __shared__ __align__(16) __bf16 aT[128 * GLD];
  __shared__ __align__(16) __bf16 bTt[64 * GLD];    // [n][k] = Wo[n0+n][k0+k]
  const int lane = threadIdx.x & 31;
  const int wave = threadIdx.x >> 5;
  const int row0 = blockIdx.x * 128;
  const int n0   = blockIdx.y * 64;

  v8f acc[2][4] = {};

  for (int k0 = 0; k0 < D_MODEL; k0 += 32) {
    __syncthreads();
    // A tile: 128x32 bf16 from ctx (row stride 1024 elems) via TDM; pad 4
    // dwords after every 16 dwords -> LDS stride 40 (interval code 3, amt 3).
#if USE_TDM
    if (wave == 0)
      tdm_load_2d_bf16((unsigned)(unsigned long long)&aT[0],
                       ctx + (size_t)row0 * D_MODEL + k0,
                       128u, 32u, (unsigned)D_MODEL, 3, 3);
#else
    {
      uint4* aTu = (uint4*)aT;
      for (int c = threadIdx.x; c < 512; c += 128) {
        int i = c >> 2, j4 = c & 3;
        aTu[i * 5 + j4] =
            ((const uint4*)(ctx + (size_t)(row0 + i) * D_MODEL + k0))[j4];
      }
    }
#endif
    // B tile: Wo rows contiguous in k -> both sides contiguous.
    {
      int n  = threadIdx.x & 63;
      int kh = (threadIdx.x >> 6) << 4;
#pragma unroll
      for (int i = 0; i < 16; ++i)
        bTt[n * GLD + kh + i] =
            (__bf16)Wo[(size_t)(n0 + n) * D_MODEL + (k0 + kh + i)];
    }
#if USE_TDM
    if (wave == 0) __builtin_amdgcn_s_wait_tensorcnt(0);
#endif
    __syncthreads();

    v16bf a0 = load_frag(&aT[(wave * 32) * GLD], GLD, lane);
    v16bf a1 = load_frag(&aT[(wave * 32 + 16) * GLD], GLD, lane);
#pragma unroll
    for (int nt = 0; nt < 4; ++nt) {
      v16bf b = load_frag(&bTt[(nt * 16) * GLD], GLD, lane);
      acc[0][nt] = wmma_bf16(a0, b, acc[0][nt]);
      acc[1][nt] = wmma_bf16(a1, b, acc[1][nt]);
    }
  }

  const int nlo = lane & 15;
#pragma unroll
  for (int rt = 0; rt < 2; ++rt) {
    const int rbase = row0 + wave * 32 + rt * 16 + ((lane >> 4) << 3);
#pragma unroll
    for (int nt = 0; nt < 4; ++nt) {
#pragma unroll
      for (int r = 0; r < 8; ++r) {
        int nc = n0 + nt * 16 + nlo;
        out[(size_t)(rbase + r) * D_MODEL + nc] = acc[rt][nt][r] + bo[nc];
      }
    }
  }
}

// ---------------------------------------------------------------------------
extern "C" void kernel_launch(void* const* d_in, const int* in_sizes, int n_in,
                              void* d_out, int out_size, void* d_ws, size_t ws_size,
                              hipStream_t stream) {
  const float* query  = (const float*)d_in[0];
  const float* key_in = (const float*)d_in[1];
  const float* value  = (const float*)d_in[2];
  const float* Wq     = (const float*)d_in[3];
  const float* Wk     = (const float*)d_in[4];
  const float* Wv     = (const float*)d_in[5];
  const float* Wo     = (const float*)d_in[6];
  const float* bo     = (const float*)d_in[7];
  float*       out    = (float*)d_out;

  const size_t nTok  = (size_t)NB * SEQ;            // 8192
  const size_t chunk = nTok * D_MODEL;
  __bf16* qbuf = (__bf16*)d_ws;                     // [B,H,S,64]
  __bf16* kbuf = qbuf + chunk;
  __bf16* vbuf = kbuf + chunk;
  __bf16* cbuf = vbuf + chunk;                      // [B,S,1024]

  dim3 blk(128);

  // Q is pre-scaled by 1/sqrt(d_model) = 1/32 (exact exponent shift in bf16).
  dim3 gProj((unsigned)(nTok / 128), NH);
  qkv_proj_kernel<<<gProj, blk, 0, stream>>>(query,  Wq, qbuf, 0.03125f);
  qkv_proj_kernel<<<gProj, blk, 0, stream>>>(key_in, Wk, kbuf, 1.0f);
  qkv_proj_kernel<<<gProj, blk, 0, stream>>>(value,  Wv, vbuf, 1.0f);

  dim3 gAttn(SEQ / 64, NB * NH);
  flash_attn_kernel<<<gAttn, blk, 0, stream>>>(qbuf, kbuf, vbuf, cbuf);

  dim3 gOut((unsigned)(nTok / 128), D_MODEL / 64);
  out_proj_kernel<<<gOut, blk, 0, stream>>>(cbuf, Wo, bo, out);
}
